// LanguageModel_72138270704057
// MI455X (gfx1250) — matrix operand order repaired and checked by reading, not compiled, not run on hardware
//
#include <hip/hip_runtime.h>

// ---------------------------------------------------------------------------
// CDNA5 (gfx1250) transformer forward:
//   bf16 WMMA GEMMs, async global->LDS double-buffered staging, f32 residual
// ---------------------------------------------------------------------------

typedef __attribute__((ext_vector_type(16))) __bf16 v16bf;
typedef __attribute__((ext_vector_type(8)))  float  v8f;
typedef __attribute__((ext_vector_type(4)))  int    v4i;

typedef __attribute__((address_space(1))) v4i  as1_v4i;
typedef __attribute__((address_space(3))) v4i  as3_v4i;
typedef __attribute__((address_space(1))) void as1_void;
typedef __attribute__((address_space(3))) void as3_void;

#define VOCAB  65
#define EMB    384
#define NHEAD  384
#define HDIM   6
#define SEQT   8
#define BATCH  2048
#define NLAYER 6
#define NTOK   (BATCH * SEQT)     // 16384
#define QKVN   (3 * NHEAD * HDIM) // 6912
#define HDTOT  (NHEAD * HDIM)     // 2304

#define DEV __device__ __forceinline__

#if defined(__has_builtin)
#if __has_builtin(__builtin_amdgcn_global_load_async_to_lds_b128)
#define HAS_ASYNC_LDS 1
#endif
#endif
#ifndef HAS_ASYNC_LDS
#define HAS_ASYNC_LDS 0
#endif

DEV float bf2f(__bf16 h) { return (float)h; }
DEV __bf16 f2bf(float f) { return (__bf16)f; }  // native v_cvt (RNE)

// 16B global -> LDS copy; async path (ASYNCcnt) when the toolchain exposes it
DEV void copy16_g2l(const void* g, void* l) {
#if HAS_ASYNC_LDS
  __builtin_amdgcn_global_load_async_to_lds_b128(
      (as1_v4i*)(as1_void*)g,
      (as3_v4i*)(as3_void*)l, 0, 0);
#else
  *(uint4*)l = *(const uint4*)g;
#endif
}

template <int N>
DEV void wait_async() {
#if HAS_ASYNC_LDS
#if __has_builtin(__builtin_amdgcn_s_wait_asynccnt)
  __builtin_amdgcn_s_wait_asynccnt((unsigned short)N);
#else
  asm volatile("s_wait_asynccnt %0" ::"i"(N));
#endif
#endif
}

// ---------------------------------------------------------------------------
// Weight packing: f32 [.., K, N] -> bf16 [N][K] (transposed, WMMA-B friendly)
// ---------------------------------------------------------------------------

__global__ __launch_bounds__(256) void pack_qkv(const float* __restrict__ wq,
                                                const float* __restrict__ wk,
                                                const float* __restrict__ wv,
                                                __bf16* __restrict__ out) {
  size_t i = (size_t)blockIdx.x * 256 + threadIdx.x;  // [L][QKVN][EMB]
  if (i >= (size_t)NLAYER * QKVN * EMB) return;
  int c = (int)(i % EMB);
  size_t r = i / EMB;
  int n = (int)(r % QKVN);
  int l = (int)(r / QKVN);
  int sec = n / HDTOT, nq = n % HDTOT, h = nq / HDIM, d = nq % HDIM;
  const float* w = (sec == 0) ? wq : (sec == 1) ? wk : wv;
  out[i] = f2bf(w[(((size_t)l * NHEAD + h) * EMB + c) * HDIM + d]);
}

__global__ __launch_bounds__(256) void pack_wo(const float* __restrict__ wo,
                                               __bf16* __restrict__ out) {
  size_t i = (size_t)blockIdx.x * 256 + threadIdx.x;  // [L][EMB][HDTOT]
  if (i >= (size_t)NLAYER * EMB * HDTOT) return;
  int k = (int)(i % HDTOT);
  size_t r = i / HDTOT;
  int n = (int)(r % EMB);
  int l = (int)(r / EMB);
  out[i] = f2bf(wo[((size_t)l * HDTOT + k) * EMB + n]);
}

__global__ __launch_bounds__(256) void pack_ff(const float* __restrict__ w,
                                               __bf16* __restrict__ out) {
  size_t i = (size_t)blockIdx.x * 256 + threadIdx.x;  // [L][EMB][EMB]
  if (i >= (size_t)NLAYER * EMB * EMB) return;
  int k = (int)(i % EMB);
  size_t r = i / EMB;
  int n = (int)(r % EMB);
  int l = (int)(r / EMB);
  out[i] = f2bf(w[((size_t)l * EMB + k) * EMB + n]);
}

__global__ __launch_bounds__(256) void pack_wout(const float* __restrict__ w,
                                                 __bf16* __restrict__ out) {
  int i = blockIdx.x * 256 + threadIdx.x;  // [128][EMB], rows >= VOCAB zeroed
  if (i >= 128 * EMB) return;
  int k = i % EMB;
  int n = i / EMB;
  out[i] = (n < VOCAB) ? f2bf(w[(size_t)k * VOCAB + n]) : f2bf(0.f);
}

// ---------------------------------------------------------------------------
// Embedding (token table reused for positions per the reference "bug");
// writes f32 residual stream + bf16 mirror for WMMA A-operand.
// ---------------------------------------------------------------------------

__global__ __launch_bounds__(256) void embed_kernel(const int* __restrict__ idx,
                                                    const float* __restrict__ tok,
                                                    float* __restrict__ x,
                                                    __bf16* __restrict__ xbf) {
  size_t i = (size_t)blockIdx.x * 256 + threadIdx.x;  // NTOK*EMB
  if (i >= (size_t)NTOK * EMB) return;
  int c = (int)(i % EMB);
  size_t tn = i / EMB;
  int t = (int)(tn & (SEQT - 1));
  float v = tok[(size_t)idx[tn] * EMB + c] + tok[(size_t)t * EMB + c];
  x[i] = v;
  xbf[i] = f2bf(v);
}

// ---------------------------------------------------------------------------
// WMMA GEMM: C[M,N] = A[M,K](bf16) * B^T (B packed [N][K] bf16) (+bias)(+relu)
// 256 thr = 8 wave32; block tile 128x64x32; wave tile 32x32 (2x2 of 16x16).
// Double-buffered LDS fed by GLOBAL_LOAD_ASYNC_TO_LDS_B128 (ASYNCcnt).
// ---------------------------------------------------------------------------

template <bool RELU, bool OUTBF16>
__global__ __launch_bounds__(256) void gemm_bf16_wmma(
    const __bf16* __restrict__ Ap, const __bf16* __restrict__ Bp,
    const float* __restrict__ bias, void* __restrict__ Cp,
    int K, int Nout, int ldc) {
  constexpr int BM = 128, BN = 64, BK = 32, SA = BK + 8, SB = BK + 8;
  __shared__ __align__(16) __bf16 As[2][BM * SA];
  __shared__ __align__(16) __bf16 Bs[2][BN * SB];

  const int tid = threadIdx.x;
  const int lane = tid & 31, wave = tid >> 5;
  const int wm = wave >> 1, wn = wave & 1;
  const int laneM = lane & 15, laneG = lane >> 4;
  const int tm0 = blockIdx.y * BM, tn0 = blockIdx.x * BN;

  const int arow = tid >> 1, aseg = tid & 1;  // A: 128 rows x 2 segs of 16
  const int brow = tid >> 2, bseg = tid & 3;  // B:  64 rows x 4 segs of 8

  const __bf16* ag0 = Ap + (size_t)(tm0 + arow) * K + aseg * 16;
  const __bf16* bg0 = Bp + (size_t)(tn0 + brow) * K + bseg * 8;
  __bf16* al[2] = { &As[0][arow * SA + aseg * 16], &As[1][arow * SA + aseg * 16] };
  __bf16* bl[2] = { &Bs[0][brow * SB + bseg * 8], &Bs[1][brow * SB + bseg * 8] };

  v8f acc[2][2];
#pragma unroll
  for (int i = 0; i < 2; i++)
#pragma unroll
    for (int j = 0; j < 2; j++)
#pragma unroll
      for (int e = 0; e < 8; e++) acc[i][j][e] = 0.0f;

  union Frag { uint4 q[2]; v16bf v; };

  const int KT = K / BK;

  // stage tile 0 into buffer 0 (3 async b128 per thread: 32B of A + 16B of B)
  copy16_g2l(ag0, al[0]);
  copy16_g2l(ag0 + 8, al[0] + 8);
  copy16_g2l(bg0, bl[0]);

  for (int kt = 0; kt < KT; kt++) {
    const int buf = kt & 1;
    if (kt + 1 < KT) {  // issue next tile into the other buffer
      const __bf16* ag = ag0 + (size_t)(kt + 1) * BK;
      const __bf16* bg = bg0 + (size_t)(kt + 1) * BK;
      copy16_g2l(ag, al[1 - buf]);
      copy16_g2l(ag + 8, al[1 - buf] + 8);
      copy16_g2l(bg, bl[1 - buf]);
      wait_async<3>();  // in-order: tile kt's 3 ops have retired
    } else {
      wait_async<0>();
    }
    __syncthreads();  // all waves landed tile kt

    // ---- fragments per ISA 7.12.2 layouts ----
    Frag a[2], b[2];
#pragma unroll
    for (int mi = 0; mi < 2; mi++) {
      int r = wm * 32 + mi * 16 + laneM;
      a[mi].q[0] = *(const uint4*)&As[buf][r * SA + laneG * 8];       // K=g*8..+7
      a[mi].q[1] = *(const uint4*)&As[buf][r * SA + 16 + laneG * 8];  // K=16+g*8..
    }
#pragma unroll
    for (int ni = 0; ni < 2; ni++) {
      int c = wn * 32 + ni * 16 + laneM;
      b[ni].q[0] = *(const uint4*)&Bs[buf][c * SB + laneG * 8];       // K=g*8..+7
      b[ni].q[1] = *(const uint4*)&Bs[buf][c * SB + 16 + laneG * 8];  // K=16+g*8..
    }
#pragma unroll
    for (int mi = 0; mi < 2; mi++)
#pragma unroll
      for (int ni = 0; ni < 2; ni++)
        acc[mi][ni] = __builtin_amdgcn_wmma_f32_16x16x32_bf16(
            false, a[mi].v, false, b[ni].v, (short)0, acc[mi][ni], false, false);
    __syncthreads();  // reads done; buffer may be overwritten next iter
  }

  // ---- epilogue ----
#pragma unroll
  for (int ni = 0; ni < 2; ni++) {
    const int col = tn0 + wn * 32 + ni * 16 + laneM;
    const bool ok = col < Nout;
    float bv = 0.f;
    if (bias != nullptr && ok) bv = bias[col];
#pragma unroll
    for (int mi = 0; mi < 2; mi++) {
      const int r0 = tm0 + wm * 32 + mi * 16 + laneG * 8;
#pragma unroll
      for (int r = 0; r < 8; r++) {
        float v = acc[mi][ni][r] + bv;
        if (RELU) v = fmaxf(v, 0.f);
        if (ok) {
          size_t o = (size_t)(r0 + r) * ldc + col;
          if (OUTBF16) ((__bf16*)Cp)[o] = f2bf(v);
          else         ((float*)Cp)[o] = v;
        }
      }
    }
  }
}

// ---------------------------------------------------------------------------
// Attention: one thread per (batch, head); T=8, D=6, causal softmax
// qkv layout per token row: [q(2304) | k(2304) | v(2304)], bf16
// ---------------------------------------------------------------------------

__global__ __launch_bounds__(256) void attention_kernel(const __bf16* __restrict__ qkv,
                                                        __bf16* __restrict__ attnc) {
  int i = blockIdx.x * 256 + threadIdx.x;
  if (i >= BATCH * NHEAD) return;
  int b = i / NHEAD, h = i % NHEAD;
  const float scale = 0.4082482904638630f;  // 6^-0.5
  size_t base = (size_t)b * SEQT * QKVN;
  size_t hc = (size_t)h * HDIM;

#pragma unroll
  for (int t = 0; t < SEQT; t++) {
    float qv[HDIM];
#pragma unroll
    for (int d = 0; d < HDIM; d++) qv[d] = bf2f(qkv[base + (size_t)t * QKVN + hc + d]);
    float w[SEQT];
    float m = -3.0e38f;
    for (int s = 0; s <= t; s++) {
      float a = 0.f;
#pragma unroll
      for (int d = 0; d < HDIM; d++)
        a += qv[d] * bf2f(qkv[base + (size_t)s * QKVN + HDTOT + hc + d]);
      a *= scale;
      w[s] = a;
      m = fmaxf(m, a);
    }
    float sum = 0.f;
    for (int s = 0; s <= t; s++) { w[s] = __expf(w[s] - m); sum += w[s]; }
    float inv = 1.f / sum;
    float o[HDIM];
#pragma unroll
    for (int d = 0; d < HDIM; d++) o[d] = 0.f;
    for (int s = 0; s <= t; s++) {
#pragma unroll
      for (int d = 0; d < HDIM; d++)
        o[d] += w[s] * bf2f(qkv[base + (size_t)s * QKVN + 2 * HDTOT + hc + d]);
    }
#pragma unroll
    for (int d = 0; d < HDIM; d++)
      attnc[(size_t)(b * SEQT + t) * HDTOT + hc + d] = f2bf(o[d] * inv);
  }
}

// ---------------------------------------------------------------------------
// x += LayerNorm(proj); refresh bf16 mirror. One wave32 per row (shfl_xor).
// ---------------------------------------------------------------------------

__global__ __launch_bounds__(256) void ln_residual(float* __restrict__ x,
                                                   __bf16* __restrict__ xbf,
                                                   const float* __restrict__ proj,
                                                   const float* __restrict__ g,
                                                   const float* __restrict__ bt) {
  int wave = threadIdx.x >> 5, lane = threadIdx.x & 31;
  int row = blockIdx.x * 8 + wave;
  const float* p = proj + (size_t)row * EMB;
  float v[EMB / 32];
  float s = 0.f, ss = 0.f;
#pragma unroll
  for (int i = 0; i < EMB / 32; i++) {
    float t = p[lane + 32 * i];
    v[i] = t; s += t; ss += t * t;
  }
#pragma unroll
  for (int o = 16; o > 0; o >>= 1) {
    s += __shfl_xor(s, o, 32);
    ss += __shfl_xor(ss, o, 32);
  }
  float mu = s * (1.f / EMB);
  float var = ss * (1.f / EMB) - mu * mu;
  float rs = rsqrtf(var + 1e-5f);
  float* xr = x + (size_t)row * EMB;
  __bf16* xb = xbf + (size_t)row * EMB;
#pragma unroll
  for (int i = 0; i < EMB / 32; i++) {
    int c = lane + 32 * i;
    float nv = xr[c] + (v[i] - mu) * rs * g[c] + bt[c];
    xr[c] = nv;
    xb[c] = f2bf(nv);
  }
}

__global__ __launch_bounds__(256) void ln_out(const float* __restrict__ x,
                                              const float* __restrict__ g,
                                              const float* __restrict__ bt,
                                              __bf16* __restrict__ out) {
  int wave = threadIdx.x >> 5, lane = threadIdx.x & 31;
  int row = blockIdx.x * 8 + wave;
  const float* p = x + (size_t)row * EMB;
  float v[EMB / 32];
  float s = 0.f, ss = 0.f;
#pragma unroll
  for (int i = 0; i < EMB / 32; i++) {
    float t = p[lane + 32 * i];
    v[i] = t; s += t; ss += t * t;
  }
#pragma unroll
  for (int o = 16; o > 0; o >>= 1) {
    s += __shfl_xor(s, o, 32);
    ss += __shfl_xor(ss, o, 32);
  }
  float mu = s * (1.f / EMB);
  float var = ss * (1.f / EMB) - mu * mu;
  float rs = rsqrtf(var + 1e-5f);
  __bf16* orow = out + (size_t)row * EMB;
#pragma unroll
  for (int i = 0; i < EMB / 32; i++) {
    int c = lane + 32 * i;
    orow[c] = f2bf((v[i] - mu) * rs * g[c] + bt[c]);
  }
}

// ---------------------------------------------------------------------------
// Cross-entropy: per-row logsumexp, then deterministic fixed-order reduction
// ---------------------------------------------------------------------------

__global__ __launch_bounds__(256) void loss_rows(const float* __restrict__ logits,
                                                 const int* __restrict__ targets,
                                                 float* __restrict__ rowloss) {
  int row = blockIdx.x * 256 + threadIdx.x;
  if (row >= NTOK) return;
  const float* l = logits + (size_t)row * VOCAB;
  float m = -3.0e38f;
  for (int j = 0; j < VOCAB; j++) m = fmaxf(m, l[j]);
  float s = 0.f;
  for (int j = 0; j < VOCAB; j++) s += __expf(l[j] - m);
  rowloss[row] = m + __logf(s) - l[targets[row]];
}

__global__ __launch_bounds__(256) void loss_reduce(const float* __restrict__ rowloss,
                                                   float* __restrict__ out) {
  __shared__ float sm[256];
  float s = 0.f;
  for (int i = threadIdx.x; i < NTOK; i += 256) s += rowloss[i];
  sm[threadIdx.x] = s;
  __syncthreads();
  for (int st = 128; st > 0; st >>= 1) {
    if ((int)threadIdx.x < st) sm[threadIdx.x] += sm[threadIdx.x + st];
    __syncthreads();
  }
  if (threadIdx.x == 0) out[0] = sm[0] * (1.f / NTOK);
}

// ---------------------------------------------------------------------------
// Host driver
// ---------------------------------------------------------------------------

extern "C" void kernel_launch(void* const* d_in, const int* in_sizes, int n_in,
                              void* d_out, int out_size, void* d_ws, size_t ws_size,
                              hipStream_t stream) {
  (void)in_sizes; (void)n_in; (void)out_size; (void)ws_size;

  const int*   idx     = (const int*)  d_in[0];
  const int*   targets = (const int*)  d_in[1];
  const float* tok     = (const float*)d_in[2];
  const float* wq      = (const float*)d_in[3];
  const float* wk      = (const float*)d_in[4];
  const float* wv      = (const float*)d_in[5];
  const float* wo      = (const float*)d_in[6];
  const float* bo      = (const float*)d_in[7];
  const float* ln1g    = (const float*)d_in[8];
  const float* ln1b    = (const float*)d_in[9];
  const float* ln2g    = (const float*)d_in[10];
  const float* ln2b    = (const float*)d_in[11];
  const float* w1      = (const float*)d_in[12];
  const float* b1      = (const float*)d_in[13];
  const float* w2      = (const float*)d_in[14];
  const float* b2      = (const float*)d_in[15];
  const float* lnog    = (const float*)d_in[16];
  const float* lnob    = (const float*)d_in[17];
  const float* woutw   = (const float*)d_in[18];
  const float* bout    = (const float*)d_in[19];

  char* ws = (char*)d_ws;
  size_t off = 0;
  auto take = [&](size_t bytes) -> char* {
    char* p = ws + off;
    off = (off + bytes + 255) & ~(size_t)255;
    return p;
  };

  float*  x     = (float*) take((size_t)NTOK * EMB * 4);
  __bf16* xbf   = (__bf16*)take((size_t)NTOK * EMB * 2);
  __bf16* qkv   = (__bf16*)take((size_t)NTOK * QKVN * 2);
  __bf16* attnc = (__bf16*)take((size_t)NTOK * HDTOT * 2);
  float*  proj  = (float*) take((size_t)NTOK * EMB * 4);
  __bf16* hbuf  = (__bf16*)take((size_t)NTOK * EMB * 2);
  __bf16* xo    = (__bf16*)take((size_t)NTOK * EMB * 2);
  __bf16* qkvw  = (__bf16*)take((size_t)NLAYER * QKVN * EMB * 2);
  __bf16* wot   = (__bf16*)take((size_t)NLAYER * EMB * HDTOT * 2);
  __bf16* w1t   = (__bf16*)take((size_t)NLAYER * EMB * EMB * 2);
  __bf16* w2t   = (__bf16*)take((size_t)NLAYER * EMB * EMB * 2);
  __bf16* woutt = (__bf16*)take((size_t)128 * EMB * 2);
  float*  rlos  = (float*) take((size_t)NTOK * 4);

  float* logits = (float*)d_out;
  float* lossp  = logits + (size_t)NTOK * VOCAB;

  // weight packing (every call; deterministic)
  pack_qkv <<<62208, 256, 0, stream>>>(wq, wk, wv, qkvw);
  pack_wo  <<<20736, 256, 0, stream>>>(wo, wot);
  pack_ff  <<<3456,  256, 0, stream>>>(w1, w1t);
  pack_ff  <<<3456,  256, 0, stream>>>(w2, w2t);
  pack_wout<<<192,   256, 0, stream>>>(woutw, woutt);

  embed_kernel<<<24576, 256, 0, stream>>>(idx, tok, x, xbf);

  for (int l = 0; l < NLAYER; l++) {
    // QKV: [16384,384] x [384,6912] -> bf16
    gemm_bf16_wmma<false, true><<<dim3(QKVN / 64, NTOK / 128), 256, 0, stream>>>(
        xbf, qkvw + (size_t)l * QKVN * EMB, nullptr, qkv, EMB, QKVN, QKVN);
    attention_kernel<<<3072, 256, 0, stream>>>(qkv, attnc);
    // Wo: [16384,2304] x [2304,384] + bo -> f32
    gemm_bf16_wmma<false, false><<<dim3(EMB / 64, NTOK / 128), 256, 0, stream>>>(
        attnc, wot + (size_t)l * EMB * HDTOT, bo + l * EMB, proj, HDTOT, EMB, EMB);
    ln_residual<<<2048, 256, 0, stream>>>(x, xbf, proj, ln1g + l * EMB, ln1b + l * EMB);
    // FFN1: relu(x@w1+b1) -> bf16
    gemm_bf16_wmma<true, true><<<dim3(EMB / 64, NTOK / 128), 256, 0, stream>>>(
        xbf, w1t + (size_t)l * EMB * EMB, b1 + l * EMB, hbuf, EMB, EMB, EMB);
    // FFN2: h@w2+b2 -> f32
    gemm_bf16_wmma<false, false><<<dim3(EMB / 64, NTOK / 128), 256, 0, stream>>>(
        hbuf, w2t + (size_t)l * EMB * EMB, b2 + l * EMB, proj, EMB, EMB, EMB);
    ln_residual<<<2048, 256, 0, stream>>>(x, xbf, proj, ln2g + l * EMB, ln2b + l * EMB);
  }

  ln_out<<<2048, 256, 0, stream>>>(x, lnog, lnob, xo);
  // logits: [16384,384] x [384,65(pad128)] + bout -> f32 into d_out (ldc=65)
  gemm_bf16_wmma<false, false><<<dim3(128 / 64, NTOK / 128), 256, 0, stream>>>(
      xo, woutt, bout, logits, EMB, VOCAB, VOCAB);

  loss_rows<<<64, 256, 0, stream>>>(logits, targets, rlos);
  loss_reduce<<<1, 256, 0, stream>>>(rlos, lossp);
}
